// GCNModel_90950227460406
// MI455X (gfx1250) — compile-verified, matrix operand order checked by reading, not run on hardware
//
#include <hip/hip_runtime.h>
#include <hip/hip_bf16.h>

// ---------------------------------------------------------------------------
// GCN stack for MI455X (gfx1250, wave32, WMMA).
//   B=8, N=10242, E=61440; widths 515 -> 512 x4 -> 64 -> 3.
// Heavy GEMMs run as bf16 WMMA (v_wmma_f32_16x16x32_bf16) with the 64-col
// weight panel staged in LDS; graph aggregation is global_atomic_add_f32.
// ---------------------------------------------------------------------------

typedef __bf16 bf16_t;
typedef __attribute__((ext_vector_type(16))) __bf16 v16bf;
typedef __attribute__((ext_vector_type(8)))  __bf16 v8bf;
typedef __attribute__((ext_vector_type(8)))  float  v8f;

#define NB   8
#define NN   10242
#define NE   61440
#define MT   (NB * NN)          // 81936 rows = 5121 strips of 16
#define GK   512                // K of every WMMA GEMM in this net

__device__ __forceinline__ void atomAddF(float* p, float v) {
  __hip_atomic_fetch_add(p, v, __ATOMIC_RELAXED, __HIP_MEMORY_SCOPE_AGENT);
}

// ---------------- degree / symmetric normalization ----------------
__global__ void k_fill1(float* p, int n) {
  int i = blockIdx.x * blockDim.x + threadIdx.x;
  if (i < n) p[i] = 1.0f;                      // self-loop contributes 1
}
__global__ void k_deg(const int* __restrict__ dst, float* deg) {
  int e = blockIdx.x * blockDim.x + threadIdx.x;
  if (e < NE) atomAddF(&deg[dst[e]], 1.0f);
}
__global__ void k_dinv(const float* __restrict__ deg, float* dinv) {
  int i = blockIdx.x * blockDim.x + threadIdx.x;
  if (i < NN) dinv[i] = rsqrtf(deg[i]);        // deg >= 1 always
}
__global__ void k_norm(const int* __restrict__ src, const int* __restrict__ dst,
                       const float* __restrict__ dinv, float* __restrict__ normv) {
  int e = blockIdx.x * blockDim.x + threadIdx.x;
  if (e < NE) normv[e] = dinv[src[e]] * dinv[dst[e]];
  else if (e < NE + NN) { float d = dinv[e - NE]; normv[e] = d * d; }
}

// ---------------- utility ----------------
__global__ void k_zero4(float4* p, size_t n4) {
  size_t i = (size_t)blockIdx.x * blockDim.x + threadIdx.x;
  size_t stride = (size_t)gridDim.x * blockDim.x;
  float4 z = {0.f, 0.f, 0.f, 0.f};
  for (; i < n4; i += stride) p[i] = z;
}
// transpose + convert weights: W[K][Fo] f32 -> Wt[Fo][K] bf16
__global__ void k_wt(const float* __restrict__ W, bf16_t* __restrict__ Wt,
                     int K, int Fo) {
  int i = blockIdx.x * blockDim.x + threadIdx.x;
  if (i < K * Fo) {
    int n = i / K, k = i - n * K;
    Wt[i] = (bf16_t)W[(size_t)k * Fo + n];
  }
}

// ---------------- layer-1 restructured (img broadcast folded) ----------------
// f1[b][o] = sum_c img[b][c] * W1[3+c][o]
__global__ void k_featproj(const float* __restrict__ img, const float* __restrict__ W1,
                           float* __restrict__ f1) {
  int b = blockIdx.x, o = threadIdx.x;          // block = 512
  float acc = 0.f;
  for (int c = 0; c < 512; ++c)
    acc = fmaf(img[b * 512 + c], W1[(size_t)(3 + c) * 512 + o], acc);
  f1[b * 512 + o] = acc;
}
// h[row][o] = v.xyz @ W1[0:3,o] + f1[b][o]
__global__ void k_h1(const float* __restrict__ verts, const float* __restrict__ W1,
                     const float* __restrict__ f1, float* __restrict__ h) {
  size_t row = blockIdx.x;
  int o = threadIdx.x;                          // block = 512
  int b = (int)(row / NN);
  float v0 = verts[row * 3 + 0], v1 = verts[row * 3 + 1], v2 = verts[row * 3 + 2];
  float acc = f1[b * 512 + o];
  acc = fmaf(v0, W1[o], acc);
  acc = fmaf(v1, W1[512 + o], acc);
  acc = fmaf(v2, W1[1024 + o], acc);
  h[row * 512 + o] = acc;
}

// ---------------- bf16 WMMA GEMM: C[M][Nn] = A[M][GK] * Bt[Nn][GK]^T ----------
// block = 256 threads (8 waves). Block tile: 256 rows x 64 cols.
// Weight panel (64 x GK bf16 = 64 KB) staged in LDS once per block.
// Wave: 2 x 16-row A strips, 4 x 16-col B tiles -> 8 WMMA per 32-wide k-step.
__global__ __launch_bounds__(256)
void k_gemm_bf16(const bf16_t* __restrict__ A, const bf16_t* __restrict__ Bt,
                 float* __restrict__ C, int M, int Nn) {
  __shared__ alignas(16) bf16_t sB[64 * GK];
  const int tid = threadIdx.x, wave = tid >> 5, lane = tid & 31;
  const int r = lane & 15, hl = lane >> 4;
  const int n0 = blockIdx.x * 64;

  {  // cooperative panel load: 256 threads x 16 x b128
    const bf16_t* src = Bt + (size_t)n0 * GK;
    for (int i = tid; i < (64 * GK) / 8; i += 256)
      *(v8bf*)(sB + (size_t)i * 8) = *(const v8bf*)(src + (size_t)i * 8);
  }
  __syncthreads();

  const int m0 = (blockIdx.y * 16 + wave * 2) * 16;
  if (m0 >= M) return;
  const int m1 = m0 + 16;
  const bool has2 = (m1 < M);
  const bf16_t* a0row = A + (size_t)(m0 + r) * GK;
  const bf16_t* a1row = A + (size_t)((has2 ? m1 : m0) + r) * GK;

  v8f acc0[4] = {}, acc1[4] = {};
  for (int k0 = 0; k0 < GK; k0 += 32) {
    union { v16bf v; v8bf h[2]; } fa0, fa1;
    fa0.h[0] = *(const v8bf*)(a0row + k0 + hl * 8);
    fa0.h[1] = *(const v8bf*)(a0row + k0 + 16 + hl * 8);
    fa1.h[0] = *(const v8bf*)(a1row + k0 + hl * 8);
    fa1.h[1] = *(const v8bf*)(a1row + k0 + 16 + hl * 8);
#pragma unroll
    for (int j = 0; j < 4; ++j) {
      union { v16bf v; v8bf h[2]; } fb;
      const bf16_t* brow = sB + (size_t)(j * 16 + r) * GK;
      fb.h[0] = *(const v8bf*)(brow + k0 + hl * 8);
      fb.h[1] = *(const v8bf*)(brow + k0 + 16 + hl * 8);
      acc0[j] = __builtin_amdgcn_wmma_f32_16x16x32_bf16(
          false, fa0.v, false, fb.v, (short)0, acc0[j], false, false);
      acc1[j] = __builtin_amdgcn_wmma_f32_16x16x32_bf16(
          false, fa1.v, false, fb.v, (short)0, acc1[j], false, false);
    }
  }
  // C/D layout: lane -> N (r) + M-half (hl), VGPR v -> M = v + 8*hl
  float* c0 = C + (size_t)m0 * Nn + n0;
#pragma unroll
  for (int j = 0; j < 4; ++j)
#pragma unroll
    for (int v = 0; v < 8; ++v)
      c0[(size_t)(v + 8 * hl) * Nn + j * 16 + r] = acc0[j][v];
  if (has2) {
    float* c1 = C + (size_t)m1 * Nn + n0;
#pragma unroll
    for (int j = 0; j < 4; ++j)
#pragma unroll
      for (int v = 0; v < 8; ++v)
        c1[(size_t)(v + 8 * hl) * Nn + j * 16 + r] = acc1[j][v];
  }
}

// ---------------- graph aggregation: o[b][dst] += norm_e * h[b][src] ---------
__global__ void k_scatter(const float* __restrict__ h, float* __restrict__ o,
                          const int* __restrict__ src, const int* __restrict__ dst,
                          const float* __restrict__ normv, int Fo) {
  int e = blockIdx.x;            // 0 .. NE+NN-1 (>= NE are self-loops)
  int b = blockIdx.y;
  int s, d;
  if (e < NE) { s = src[e]; d = dst[e]; } else { s = d = e - NE; }
  float w = normv[e];
  const float* hr = h + ((size_t)b * NN + s) * (size_t)Fo;
  float* orow     = o + ((size_t)b * NN + d) * (size_t)Fo;
  for (int f = threadIdx.x; f < Fo; f += blockDim.x)
    atomAddF(&orow[f], w * hr[f]);
}

// ---------------- epilogues ----------------
__global__ void k_epi_bf16(const float* __restrict__ o, const float* __restrict__ bias,
                           bf16_t* __restrict__ xb, int Fo, size_t total, int relu) {
  size_t i = (size_t)blockIdx.x * blockDim.x + threadIdx.x;
  size_t stride = (size_t)gridDim.x * blockDim.x;
  for (; i < total; i += stride) {
    float v = o[i] + bias[i % (size_t)Fo];
    if (relu) v = fmaxf(v, 0.f);
    xb[i] = (bf16_t)v;
  }
}
__global__ void k_epi_out(const float* __restrict__ o, const float* __restrict__ bias,
                          float* __restrict__ out, size_t total) {
  size_t i = (size_t)blockIdx.x * blockDim.x + threadIdx.x;
  size_t stride = (size_t)gridDim.x * blockDim.x;
  for (; i < total; i += stride) out[i] = o[i] + bias[i % 3];
}

// ---------------- tiny final GEMM: 64 -> 3 ----------------
__global__ void k_gemm6(const bf16_t* __restrict__ X, const float* __restrict__ W6,
                        float* __restrict__ h) {
  int row = blockIdx.x * blockDim.x + threadIdx.x;
  if (row >= MT) return;
  const bf16_t* xr = X + (size_t)row * 64;
  float s0 = 0.f, s1 = 0.f, s2 = 0.f;
  for (int k = 0; k < 64; ++k) {
    float x = (float)xr[k];
    s0 = fmaf(x, W6[k * 3 + 0], s0);
    s1 = fmaf(x, W6[k * 3 + 1], s1);
    s2 = fmaf(x, W6[k * 3 + 2], s2);
  }
  h[(size_t)row * 3 + 0] = s0;
  h[(size_t)row * 3 + 1] = s1;
  h[(size_t)row * 3 + 2] = s2;
}

// ---------------- host-side layer tail: zero, scatter, epilogue --------------
static void gcn_tail(const float* h, float* o, const float* bias,
                     bf16_t* xbOut, float* fOut, int Fo, int relu,
                     const int* srcI, const int* dstI, const float* normv,
                     hipStream_t stream) {
  size_t total = (size_t)MT * Fo;
  k_zero4<<<dim3(2048), 256, 0, stream>>>((float4*)o, total / 4);
  int blk = Fo >= 256 ? 256 : (Fo >= 64 ? 64 : 32);
  k_scatter<<<dim3(NE + NN, NB), blk, 0, stream>>>(h, o, srcI, dstI, normv, Fo);
  if (xbOut)
    k_epi_bf16<<<dim3(4096), 256, 0, stream>>>(o, bias, xbOut, Fo, total, relu);
  else
    k_epi_out<<<dim3(2048), 256, 0, stream>>>(o, bias, fOut, total);
}

extern "C" void kernel_launch(void* const* d_in, const int* in_sizes, int n_in,
                              void* d_out, int out_size, void* d_ws, size_t ws_size,
                              hipStream_t stream) {
  (void)in_sizes; (void)n_in; (void)out_size; (void)ws_size;
  const float* verts = (const float*)d_in[0];
  const float* img   = (const float*)d_in[1];
  const int*   ei    = (const int*)d_in[2];
  const float* W1 = (const float*)d_in[3];
  const float* b1 = (const float*)d_in[4];
  const float* W2 = (const float*)d_in[5];
  const float* b2 = (const float*)d_in[6];
  const float* W3 = (const float*)d_in[7];
  const float* b3 = (const float*)d_in[8];
  const float* W4 = (const float*)d_in[9];
  const float* b4 = (const float*)d_in[10];
  const float* W5 = (const float*)d_in[11];
  const float* b5 = (const float*)d_in[12];
  const float* W6 = (const float*)d_in[13];
  const float* b6 = (const float*)d_in[14];
  float* out = (float*)d_out;

  // workspace carve-up (~421 MB), 256B aligned
  char* ws = (char*)d_ws;
  size_t off = 0;
  auto take = [&](size_t bytes) -> char* {
    char* p = ws + off;
    off = (off + bytes + 255) & ~(size_t)255;
    return p;
  };
  float*  o_f   = (float*) take((size_t)MT * 512 * 4);   // scatter accumulator
  float*  h_f   = (float*) take((size_t)MT * 512 * 4);   // GEMM output
  bf16_t* xb    = (bf16_t*)take((size_t)MT * 512 * 2);   // bf16 activations
  bf16_t* wt    = (bf16_t*)take((size_t)512 * 512 * 2);  // transposed weights
  float*  f1    = (float*) take((size_t)NB * 512 * 4);
  float*  deg   = (float*) take((size_t)NN * 4);
  float*  dinv  = (float*) take((size_t)NN * 4);
  float*  normv = (float*) take((size_t)(NE + NN) * 4);

  const int* srcI = ei;        // edge_indices[0]
  const int* dstI = ei + NE;   // edge_indices[1]

  // degree / normalization (recomputed every call; deterministic)
  k_fill1<<<(NN + 255) / 256, 256, 0, stream>>>(deg, NN);
  k_deg  <<<(NE + 255) / 256, 256, 0, stream>>>(dstI, deg);
  k_dinv <<<(NN + 255) / 256, 256, 0, stream>>>(deg, dinv);
  k_norm <<<(NE + NN + 255) / 256, 256, 0, stream>>>(srcI, dstI, dinv, normv);

  // ---- layer 1 (restructured: broadcast img folded into per-batch GEMV) ----
  k_featproj<<<NB, 512, 0, stream>>>(img, W1, f1);
  k_h1<<<MT, 512, 0, stream>>>(verts, W1, f1, h_f);
  gcn_tail(h_f, o_f, b1, xb, nullptr, 512, 0, srcI, dstI, normv, stream);

  const dim3 gg(512 / 64, (MT / 16 + 15) / 16);   // (8, 321)
  const dim3 g5(1,        (MT / 16 + 15) / 16);   // (1, 321)

  // ---- layer 2 (512x512 WMMA) + ReLU ----
  k_wt<<<(512 * 512 + 255) / 256, 256, 0, stream>>>(W2, wt, 512, 512);
  k_gemm_bf16<<<gg, 256, 0, stream>>>(xb, wt, h_f, MT, 512);
  gcn_tail(h_f, o_f, b2, xb, nullptr, 512, 1, srcI, dstI, normv, stream);

  // ---- layer 3 ----
  k_wt<<<(512 * 512 + 255) / 256, 256, 0, stream>>>(W3, wt, 512, 512);
  k_gemm_bf16<<<gg, 256, 0, stream>>>(xb, wt, h_f, MT, 512);
  gcn_tail(h_f, o_f, b3, xb, nullptr, 512, 0, srcI, dstI, normv, stream);

  // ---- layer 4 + ReLU ----
  k_wt<<<(512 * 512 + 255) / 256, 256, 0, stream>>>(W4, wt, 512, 512);
  k_gemm_bf16<<<gg, 256, 0, stream>>>(xb, wt, h_f, MT, 512);
  gcn_tail(h_f, o_f, b4, xb, nullptr, 512, 1, srcI, dstI, normv, stream);

  // ---- layer 5 (512 -> 64 WMMA) ----
  k_wt<<<(512 * 64 + 255) / 256, 256, 0, stream>>>(W5, wt, 512, 64);
  k_gemm_bf16<<<g5, 256, 0, stream>>>(xb, wt, h_f, MT, 64);
  gcn_tail(h_f, o_f, b5, xb, nullptr, 64, 0, srcI, dstI, normv, stream);

  // ---- layer 6 (64 -> 3, scalar) ----
  k_gemm6<<<(MT + 255) / 256, 256, 0, stream>>>(xb, W6, h_f);
  gcn_tail(h_f, o_f, b6, nullptr, out, 3, 0, srcI, dstI, normv, stream);
}
